// GraphConvolution_71863392796808
// MI455X (gfx1250) — compile-verified, matrix operand order checked by reading, not run on hardware
//
#include <hip/hip_runtime.h>
#include <hip/hip_bf16.h>

// ---------------------------------------------------------------------------
// GCN layer for MI455X (gfx1250, wave32, WMMA):
//   support = x @ W ; out = adj @ support + bias
// bf16 WMMA 16x16x32, f32 accumulate. adj (134MB) dominates HBM traffic.
// This revision: the block's 32x32 adj tile is staged through double-buffered
// LDS (f32 -> bf16 once per block, not once per wave), removing the 4x
// redundant adj fetch; LDS rows padded to 80B for conflict-free ds_load_b128.
// One __syncthreads per k-step (double buffering covers both hazards).
// ---------------------------------------------------------------------------

typedef __attribute__((ext_vector_type(8)))  float          v8f;
typedef __attribute__((ext_vector_type(4)))  float          v4f;
typedef __attribute__((ext_vector_type(16))) __bf16         v16bf;
typedef __attribute__((ext_vector_type(4)))  unsigned int   v4u;
typedef __attribute__((ext_vector_type(2)))  unsigned int   v2u;

#define BATCH 8
#define NN    2048
#define FF    256
#define ASTRIDE 40   // ushorts per LDS A-row (80B; 20 dwords -> bank-conflict-free)

// pack two f32 -> dword of two bf16 (RNE). Builtin not exposed for gfx1250;
// the instruction exists, emit directly.
__device__ __forceinline__ unsigned int pk_bf16(float lo, float hi) {
  unsigned int d;
  asm("v_cvt_pk_bf16_f32 %0, %1, %2" : "=v"(d) : "v"(lo), "v"(hi));
  return d;
}

__device__ __forceinline__ unsigned short f2bf(float f) {
  unsigned int u = __builtin_bit_cast(unsigned int, f);
  return (unsigned short)((u + 0x7FFFu + ((u >> 16) & 1u)) >> 16);
}

union Frag { v16bf bf; v4u u4[2]; };

struct RawA { v4f r0, r1, r2, r3; };

__device__ __forceinline__ RawA load_a_raw(const float* __restrict__ row, int kb) {
  RawA ra;
  ra.r0 = *(const v4f*)(row + kb);
  ra.r1 = *(const v4f*)(row + kb + 4);
  ra.r2 = *(const v4f*)(row + kb + 16);
  ra.r3 = *(const v4f*)(row + kb + 20);
  return ra;
}

__device__ __forceinline__ v16bf cvt_a(const RawA& ra) {
  Frag f;
  f.u4[0][0] = pk_bf16(ra.r0[0], ra.r0[1]);
  f.u4[0][1] = pk_bf16(ra.r0[2], ra.r0[3]);
  f.u4[0][2] = pk_bf16(ra.r1[0], ra.r1[1]);
  f.u4[0][3] = pk_bf16(ra.r1[2], ra.r1[3]);
  f.u4[1][0] = pk_bf16(ra.r2[0], ra.r2[1]);
  f.u4[1][1] = pk_bf16(ra.r2[2], ra.r2[3]);
  f.u4[1][2] = pk_bf16(ra.r3[0], ra.r3[1]);
  f.u4[1][3] = pk_bf16(ra.r3[2], ra.r3[3]);
  return f.bf;
}

// B fragment (32x16 bf16): lane L holds column N=L; contiguous 16B loads.
__device__ __forceinline__ v16bf load_b_bf16(const unsigned short* __restrict__ col, int kb) {
  Frag f;
  f.u4[0] = *(const v4u*)(col + kb);
  f.u4[1] = *(const v4u*)(col + kb + 16);
  return f.bf;
}

// A fragment (16x32 bf16) from the LDS-staged tile.
__device__ __forceinline__ v16bf load_a_lds(const unsigned short* buf, int row, int lh8) {
  Frag f;
  f.u4[0] = *(const v4u*)(buf + row * ASTRIDE + lh8);
  f.u4[1] = *(const v4u*)(buf + row * ASTRIDE + lh8 + 16);
  return f.bf;
}

// ---- kernel 0: Wt[o][i] = bf16(W[i][o]) ------------------------------------
__global__ void prep_wt(const float* __restrict__ w, unsigned short* __restrict__ wt) {
  int i = blockIdx.x;
  int o = threadIdx.x;
  wt[(size_t)o * FF + i] = f2bf(w[(size_t)i * FF + o]);
}

// ---- kernel 1: supportT[b][o][m] = bf16( sum_i x[b][m][i] * W[i][o] ) ------
__global__ void __launch_bounds__(256)
gemm_support(const float* __restrict__ x, const unsigned short* __restrict__ wt,
             unsigned short* __restrict__ supT) {
  const int b     = blockIdx.y;
  const int lane  = threadIdx.x & 31;
  const int wave  = threadIdx.x >> 5;
  const int m0    = blockIdx.x * 32 + (wave >> 2) * 16;
  const int o0    = (wave & 3) * 64;
  const int lhalf = lane >> 4;
  const int l15   = lane & 15;
  const int lh8   = lhalf * 8;

  const float* xrow = x + ((size_t)b * NN + (m0 + l15)) * FF;
  const unsigned short* bcol0 = wt + (size_t)(o0 + l15) * FF;

  v8f c[4] = {};
  RawA ra = load_a_raw(xrow, lh8);
  v16bf bcur[4];
#pragma unroll
  for (int t = 0; t < 4; ++t) bcur[t] = load_b_bf16(bcol0 + (size_t)t * 16 * FF, lh8);

#pragma unroll 2
  for (int k = 0; k < FF - 32; k += 32) {
    const int kb = k + 32 + lh8;
    RawA ran = load_a_raw(xrow, kb);
    v16bf bn[4];
#pragma unroll
    for (int t = 0; t < 4; ++t) bn[t] = load_b_bf16(bcol0 + (size_t)t * 16 * FF, kb);
    v16bf a = cvt_a(ra);
#pragma unroll
    for (int t = 0; t < 4; ++t)
      c[t] = __builtin_amdgcn_wmma_f32_16x16x32_bf16(false, a, false, bcur[t],
                                                     (short)0, c[t], false, false);
    ra = ran;
#pragma unroll
    for (int t = 0; t < 4; ++t) bcur[t] = bn[t];
  }
  {
    v16bf a = cvt_a(ra);
#pragma unroll
    for (int t = 0; t < 4; ++t)
      c[t] = __builtin_amdgcn_wmma_f32_16x16x32_bf16(false, a, false, bcur[t],
                                                     (short)0, c[t], false, false);
  }

#pragma unroll
  for (int t = 0; t < 4; ++t) {
    unsigned short* dst =
        supT + ((size_t)b * FF + (o0 + t * 16 + l15)) * NN + m0 + lh8;
    v4u pk;
#pragma unroll
    for (int r = 0; r < 4; ++r) pk[r] = pk_bf16(c[t][2 * r], c[t][2 * r + 1]);
    *(v4u*)dst = pk;
  }
}

// ---- kernel 2: out[b][n][o] = sum_m adj[b][n][m] * supportT[b][o][m] + bias[o]
// Block = 8 waves, 32 n-rows x 256 o-cols. adj tile staged via LDS (bf16).
__global__ void __launch_bounds__(256)
gemm_aggregate(const float* __restrict__ adj, const unsigned short* __restrict__ supT,
               const float* __restrict__ bias, float* __restrict__ out) {
  __shared__ unsigned short lsA[2][32 * ASTRIDE];   // 2 x 2560B

  const int tid   = threadIdx.x;
  const int b     = blockIdx.y;
  const int nblk  = blockIdx.x * 32;
  const int lane  = tid & 31;
  const int wave  = tid >> 5;
  const int strip = wave >> 2;            // 0/1 -> n-rows strip
  const int o0    = (wave & 3) * 64;
  const int lhalf = lane >> 4;
  const int l15   = lane & 15;
  const int lh8   = lhalf * 8;
  const int arow  = strip * 16 + l15;     // this lane's A row in the LDS tile

  // staging assignment: thread -> (row = tid>>3, 4 floats at col (tid&7)*4)
  const int srow = tid >> 3;
  const int scol = (tid & 7) * 4;
  const float* astage = adj + ((size_t)b * NN + nblk + srow) * NN + scol;
  unsigned short* sdst0 = &lsA[0][srow * ASTRIDE + scol];
  unsigned short* sdst1 = &lsA[1][srow * ASTRIDE + scol];

  const unsigned short* bcol0 = supT + ((size_t)b * FF + (o0 + l15)) * NN;

  // prologue: stage tile k=0 into buf0, preload B for k=0
  {
    v4f r = *(const v4f*)astage;
    v2u p;
    p[0] = pk_bf16(r[0], r[1]);
    p[1] = pk_bf16(r[2], r[3]);
    *(v2u*)sdst0 = p;
  }
  v16bf bcur[4];
#pragma unroll
  for (int t = 0; t < 4; ++t) bcur[t] = load_b_bf16(bcol0 + (size_t)t * 16 * NN, lh8);
  __syncthreads();

  v8f c[4] = {};
  const int NIT = NN / 32;                // 64 k-steps
#pragma unroll 1
  for (int it = 0; it < NIT - 1; ++it) {
    const int cur = it & 1;
    // issue next tile's staging load + next B loads before consuming current
    v4f r = *(const v4f*)(astage + (it + 1) * 32);
    v16bf bn[4];
    const int kbn = (it + 1) * 32 + lh8;
#pragma unroll
    for (int t = 0; t < 4; ++t) bn[t] = load_b_bf16(bcol0 + (size_t)t * 16 * NN, kbn);
    // prefetch adj HBM stream ~16 k-steps ahead
    if (it + 17 <= NIT) __builtin_prefetch(astage + (it + 16) * 32, 0, 1);

    // consume current LDS tile
    v16bf a = load_a_lds(lsA[cur], arow, lh8);
#pragma unroll
    for (int t = 0; t < 4; ++t)
      c[t] = __builtin_amdgcn_wmma_f32_16x16x32_bf16(false, a, false, bcur[t],
                                                     (short)0, c[t], false, false);

    // convert + store next tile into the other buffer
    {
      v2u p;
      p[0] = pk_bf16(r[0], r[1]);
      p[1] = pk_bf16(r[2], r[3]);
      *(v2u*)(cur ? sdst0 : sdst1) = p;
    }
#pragma unroll
    for (int t = 0; t < 4; ++t) bcur[t] = bn[t];
    __syncthreads();
  }
  // peeled last k-step
  {
    v16bf a = load_a_lds(lsA[(NIT - 1) & 1], arow, lh8);
#pragma unroll
    for (int t = 0; t < 4; ++t)
      c[t] = __builtin_amdgcn_wmma_f32_16x16x32_bf16(false, a, false, bcur[t],
                                                     (short)0, c[t], false, false);
  }

  // Epilogue: bias add + f32 stores. C VGPR r -> row nblk + strip*16 + lh8 + r.
#pragma unroll
  for (int t = 0; t < 4; ++t) {
    const int col = o0 + t * 16 + l15;
    const float bv = bias[col];
    float* dst = out + ((size_t)b * NN + nblk + strip * 16 + lh8) * FF + col;
#pragma unroll
    for (int r = 0; r < 8; ++r) dst[(size_t)r * FF] = c[t][r] + bv;
  }
}

extern "C" void kernel_launch(void* const* d_in, const int* in_sizes, int n_in,
                              void* d_out, int out_size, void* d_ws, size_t ws_size,
                              hipStream_t stream) {
  const float* x    = (const float*)d_in[0];
  const float* adj  = (const float*)d_in[1];
  const float* w    = (const float*)d_in[2];
  const float* bias = (const float*)d_in[3];
  float* out = (float*)d_out;

  unsigned short* wt   = (unsigned short*)d_ws;          // 128 KB
  unsigned short* supT = wt + (size_t)FF * FF;           // 8 MB

  prep_wt<<<dim3(FF), dim3(FF), 0, stream>>>(w, wt);
  gemm_support<<<dim3(NN / 32, BATCH), dim3(256), 0, stream>>>(x, wt, supT);
  gemm_aggregate<<<dim3(NN / 32, BATCH), dim3(256), 0, stream>>>(adj, supT, bias, out);
}
